// DenseCRFLoss_38817914421724
// MI455X (gfx1250) — matrix operand order repaired and checked by reading, not compiled, not run on hardware
//
#include <hip/hip_runtime.h>

typedef __attribute__((ext_vector_type(16))) _Float16 v16h;
typedef __attribute__((ext_vector_type(8)))  _Float16 v8h;
typedef __attribute__((ext_vector_type(8)))  float    v8f;

#define Hh 96
#define Wh 96
#define PTS (Hh*Wh)          // 9216 points per image after downsample
#define B_IMG 4
#define K_SEG 21
#define NT (PTS/16)          // 576 tiles of 16 points
#define JCHUNKS 8
#define JT_PER (NT/JCHUNKS)  // 72 j-tiles per wave

// -0.5 * log2(e): exp(-0.5*d2) == exp2(EXP2_SCALE * d2)
#define EXP2_SCALE (-0.72134752044448170f)
// loss scale: WEIGHT * (-1/B) = 2e-9 * (-0.25)
#define LOSS_SCALE (-5.0e-10f)

__device__ __forceinline__ float fast_exp2(float x) {
#if __has_builtin(__builtin_amdgcn_exp2f)
  return __builtin_amdgcn_exp2f(x);   // v_exp_f32
#else
  return exp2f(x);
#endif
}

// ---------------------------------------------------------------------------
// Prep: build per-point WMMA-friendly rows (32 f16, zero padded) for the
// 5-dim bilateral features and the 21-dim segmentation vectors, plus |f|^2.
// ---------------------------------------------------------------------------
__global__ void crf_prep(const float* __restrict__ seg,   // [4,21,192,192]
                         const float* __restrict__ img,   // [4,3,192,192]
                         _Float16* __restrict__ fpad,     // [4*PTS, 32]
                         _Float16* __restrict__ Sv,       // [4*PTS, 32]
                         float* __restrict__ sq)          // [4*PTS]
{
  int t = blockIdx.x * blockDim.x + threadIdx.x;
  if (t >= B_IMG * PTS) return;
  int b = t / PTS, p = t % PTS;
  int y = p / Wh, x = p % Wh;
  int off = (2 * y) * 192 + 2 * x;   // nearest-downsample source offset

  float f[5];
  f[0] = (float)x * (1.0f / 50.0f);  // sigma_xy * scale = 50
  f[1] = (float)y * (1.0f / 50.0f);
  const float* ib = img + (size_t)b * 3 * 192 * 192;
  f[2] = ib[0 * 192 * 192 + off] * (1.0f / 15.0f);
  f[3] = ib[1 * 192 * 192 + off] * (1.0f / 15.0f);
  f[4] = ib[2 * 192 * 192 + off] * (1.0f / 15.0f);

  float s2 = 0.f;
  _Float16* fr = fpad + (size_t)t * 32;
#pragma unroll
  for (int k = 0; k < 5; ++k) { s2 += f[k] * f[k]; fr[k] = (_Float16)f[k]; }
#pragma unroll
  for (int k = 5; k < 32; ++k) fr[k] = (_Float16)0.f;
  sq[t] = s2;

  // bilinear half-downsample of segmentations == 2x2 average pool
  _Float16* sr = Sv + (size_t)t * 32;
  const float* sb = seg + (size_t)b * K_SEG * 192 * 192;
#pragma unroll
  for (int k = 0; k < K_SEG; ++k) {
    const float* sk = sb + (size_t)k * 192 * 192 + off;
    float v = 0.25f * (sk[0] + sk[1] + sk[192] + sk[193]);
    sr[k] = (_Float16)v;
  }
#pragma unroll
  for (int k = K_SEG; k < 32; ++k) sr[k] = (_Float16)0.f;
}

// ---------------------------------------------------------------------------
// Main: one wave owns (image b, i-tile, j-chunk). Two v_wmma_f32_16x16x32_f16
// per tile pair (feature Gram + segmentation Gram), fused exp + accumulate.
// ---------------------------------------------------------------------------
__global__ void crf_main(const _Float16* __restrict__ fpad,
                         const _Float16* __restrict__ Sv,
                         const float* __restrict__ sq,
                         float* __restrict__ out)
{
  int wave = (blockIdx.x * blockDim.x + threadIdx.x) >> 5;
  int lane = threadIdx.x & 31;
  int jc  = wave % JCHUNKS;
  int rem = wave / JCHUNKS;
  int ti  = rem % NT;
  int b   = rem / NT;

  const _Float16* fb  = fpad + (size_t)b * PTS * 32;
  const _Float16* sb  = Sv   + (size_t)b * PTS * 32;
  const float*    sqb = sq   + (size_t)b * PTS;

  int  lh = lane & 15;
  bool hi = lane >= 16;

  // --- A operands (i-tile), invariant over the j loop ---
  // A 16x32 f16, contiguous k order used here (B matches): lanes 0-15 hold K0..15; lanes 16-31 K16..31
  int pi = ti * 16 + lh;
  const _Float16* fa = fb + (size_t)pi * 32 + (hi ? 16 : 0);
  v8h a0 = *(const v8h*)(fa);
  v8h a1 = *(const v8h*)(fa + 8);
  v16h aF = __builtin_shufflevector(a0, a1, 0,1,2,3,4,5,6,7,8,9,10,11,12,13,14,15);
  const _Float16* sa = sb + (size_t)pi * 32 + (hi ? 16 : 0);
  v8h s0 = *(const v8h*)(sa);
  v8h s1 = *(const v8h*)(sa + 8);
  v16h aS = __builtin_shufflevector(s0, s1, 0,1,2,3,4,5,6,7,8,9,10,11,12,13,14,15);

  // |f_i|^2 for the 8 accumulator rows this lane covers:
  // C/D layout: VGPR r holds M = r + (lane>=16 ? 8 : 0), N = lane&15
  float sqi[8];
  int mbase = ti * 16 + (hi ? 8 : 0);
#pragma unroll
  for (int r = 0; r < 8; ++r) sqi[r] = sqb[mbase + r];

  float acc = 0.f;
  int tj0 = jc * JT_PER;
  for (int tj = tj0; tj < tj0 + JT_PER; ++tj) {
    // B 32x16 f16, same contiguous k order as A: lanes 0-15 hold K0..15 of column n=lane;
    // lanes 16-31 hold K16..31 of column n=lane-16  -> contiguous 32B chunk
    int pj = tj * 16 + lh;
    v16h bF = *(const v16h*)(fb + (size_t)pj * 32 + (hi ? 16 : 0));
    v16h bS = *(const v16h*)(sb + (size_t)pj * 32 + (hi ? 16 : 0));
    float sqj = sqb[pj];

    v8f zero = {};
    v8f dot = __builtin_amdgcn_wmma_f32_16x16x32_f16(
        false, aF, false, bF, (short)0, zero, false, false);
    v8f g = __builtin_amdgcn_wmma_f32_16x16x32_f16(
        false, aS, false, bS, (short)0, zero, false, false);

#pragma unroll
    for (int r = 0; r < 8; ++r) {
      float d2 = sqi[r] + sqj - 2.0f * dot[r];
      acc += g[r] * fast_exp2(EXP2_SCALE * d2);
    }
  }

  // wave32 reduction, then one scaled atomic per wave
  for (int offd = 16; offd > 0; offd >>= 1)
    acc += __shfl_xor(acc, offd, 32);
  if (lane == 0)
    atomicAdd(out, acc * LOSS_SCALE);
}

extern "C" void kernel_launch(void* const* d_in, const int* in_sizes, int n_in,
                              void* d_out, int out_size, void* d_ws, size_t ws_size,
                              hipStream_t stream) {
  const float* seg = (const float*)d_in[0];  // [4,21,192,192]
  const float* img = (const float*)d_in[1];  // [4,3,192,192]

  char* ws = (char*)d_ws;
  const size_t row_bytes = (size_t)B_IMG * PTS * 32 * sizeof(_Float16); // 2.36 MB
  _Float16* fpad = (_Float16*)ws;
  _Float16* Svec = (_Float16*)(ws + row_bytes);
  float*    sqv  = (float*)(ws + 2 * row_bytes);

  float* out = (float*)d_out;
  hipMemsetAsync(out, 0, sizeof(float), stream);

  int prep_threads = B_IMG * PTS;                      // 36864
  crf_prep<<<(prep_threads + 255) / 256, 256, 0, stream>>>(seg, img, fpad, Svec, sqv);

  int waves = B_IMG * NT * JCHUNKS;                    // 18432 waves
  crf_main<<<waves / 8, 256, 0, stream>>>(fpad, Svec, sqv, out);
}
